// SelfAttention2d_6700148982291
// MI455X (gfx1250) — matrix operand order repaired and checked by reading, not compiled, not run on hardware
//
#include <hip/hip_runtime.h>

typedef __attribute__((ext_vector_type(16))) __bf16 v16bf;
typedef __attribute__((ext_vector_type(8)))  __bf16 v8bf;
typedef __attribute__((ext_vector_type(8)))  float  v8f;

#define CC 256
#define C8 32
#define NN 4096
#define BB 8

__device__ __forceinline__ unsigned short f2bf_bits(float f) {
  unsigned u = __builtin_bit_cast(unsigned, f);
  u = (u + 0x7FFFu + ((u >> 16) & 1u)) >> 16;  // round-to-nearest-even
  return (unsigned short)u;
}

__device__ __forceinline__ v8bf ld8(const unsigned short* p) {
  v8bf r;
  __builtin_memcpy(&r, p, 16);  // -> b128 load (global or ds)
  return r;
}
__device__ __forceinline__ v16bf cat(v8bf lo, v8bf hi) {
  return __builtin_shufflevector(lo, hi, 0, 1, 2, 3, 4, 5, 6, 7, 8, 9, 10, 11, 12, 13, 14, 15);
}
__device__ __forceinline__ v8f wmma_bf16(v16bf a, v16bf b, v8f c) {
  return __builtin_amdgcn_wmma_f32_16x16x32_bf16(false, a, false, b, (short)0, c, false, false);
}

// Async global->LDS copy, 16B per lane (ASYNCcnt-tracked, GVS addressing).
__device__ __forceinline__ void async_copy_b128(unsigned lds_off,
                                                const unsigned short* gbase,
                                                unsigned gofs) {
  asm volatile("global_load_async_to_lds_b128 %0, %1, %2"
               :: "v"(lds_off), "v"(gofs), "s"(gbase)
               : "memory");
}

// ---------------------------------------------------------------------------
// x[b][c][n] (f32) -> xt[b][n][c] (bf16), LDS-tiled transpose, coalesced both ways
// ---------------------------------------------------------------------------
__global__ __launch_bounds__(256) void transpose_x_bf16(
    const float* __restrict__ x, unsigned short* __restrict__ xt) {
  __shared__ unsigned short tile[32][33];
  const int nBase = blockIdx.x * 32;
  const int cBase = blockIdx.y * 32;
  const int b     = blockIdx.z;
  const float* xb = x + (size_t)b * CC * NN;
  unsigned short* xtb = xt + (size_t)b * NN * CC;
#pragma unroll
  for (int j = 0; j < 4; ++j) {
    int cL = threadIdx.y + j * 8;
    tile[cL][threadIdx.x] = f2bf_bits(xb[(size_t)(cBase + cL) * NN + nBase + threadIdx.x]);
  }
  __syncthreads();
#pragma unroll
  for (int j = 0; j < 4; ++j) {
    int nL = threadIdx.y + j * 8;
    xtb[(size_t)(nBase + nL) * CC + cBase + threadIdx.x] = tile[threadIdx.x][nL];
  }
}

__global__ __launch_bounds__(256) void cvt_bf16(const float* __restrict__ s,
                                                unsigned short* __restrict__ d, int n) {
  int i = blockIdx.x * 256 + threadIdx.x;
  if (i < n) d[i] = f2bf_bits(s[i]);
}

// ---------------------------------------------------------------------------
// Shared WMMA mainloop: acc[16x16] = W[16x256] * X^T[256x16]  (all bf16 operands)
// ---------------------------------------------------------------------------
__device__ __forceinline__ v8f proj_mainloop(const unsigned short* __restrict__ wb,
                                             const unsigned short* __restrict__ xtb,
                                             int mBase, int nBase, int half, int l16) {
  v8f acc = {};
  const unsigned short* wr = wb + (size_t)(mBase + l16) * CC + half * 8;
  const unsigned short* xr = xtb + (size_t)(nBase + l16) * CC + half * 8;
#pragma unroll
  for (int kk = 0; kk < CC; kk += 32) {
    v16bf a  = cat(ld8(wr + kk), ld8(wr + kk + 16));
    v16bf bm = cat(ld8(xr + kk), ld8(xr + kk + 16));
    acc = wmma_bf16(a, bm, acc);
  }
  return acc;
}

// Q/K projection with transposed output: outT[b][n][m] (row length 32).
__global__ __launch_bounds__(32) void proj_qk_t(
    const unsigned short* __restrict__ xt, const unsigned short* __restrict__ wb,
    const float* __restrict__ bias, unsigned short* __restrict__ outT) {
  const int lane = threadIdx.x, half = lane >> 4, l16 = lane & 15;
  const int nBase = blockIdx.x * 16, mBase = blockIdx.y * 16, b = blockIdx.z;
  const unsigned short* xtb = xt + (size_t)b * NN * CC;
  v8f acc = proj_mainloop(wb, xtb, mBase, nBase, half, l16);
  unsigned short tmp[8];
#pragma unroll
  for (int r = 0; r < 8; ++r)
    tmp[r] = f2bf_bits(acc[r] + bias[mBase + 8 * half + r]);
  __builtin_memcpy(outT + (size_t)b * NN * C8 + (size_t)(nBase + l16) * C8 + mBase + 8 * half,
                   tmp, 16);
}

// V projection, output V[b][c][n] (n contiguous -> contiguous B-operand for P*V)
__global__ __launch_bounds__(32) void proj_v(
    const unsigned short* __restrict__ xt, const unsigned short* __restrict__ wb,
    const float* __restrict__ bias, unsigned short* __restrict__ outb) {
  const int lane = threadIdx.x, half = lane >> 4, l16 = lane & 15;
  const int nBase = blockIdx.x * 16, mBase = blockIdx.y * 16, b = blockIdx.z;
  const unsigned short* xtb = xt + (size_t)b * NN * CC;
  unsigned short* ob = outb + (size_t)b * CC * NN;
  v8f acc = proj_mainloop(wb, xtb, mBase, nBase, half, l16);
#pragma unroll
  for (int r = 0; r < 8; ++r) {
    int c = mBase + r + 8 * half;
    ob[(size_t)c * NN + nBase + l16] = f2bf_bits(acc[r] + bias[c]);
  }
}

// ---------------------------------------------------------------------------
// Fused flash attention: 2 waves / block; 16-query block; 64-key inner block.
// K tiles (4KB, shared by both waves) are double-buffered in LDS via
// global_load_async_to_lds_b128; next V block is prefetched each iteration.
// ---------------------------------------------------------------------------
__global__ __launch_bounds__(64) void flash_attn(
    const unsigned short* __restrict__ Qt, const unsigned short* __restrict__ Kt,
    const unsigned short* __restrict__ V, const float* __restrict__ x,
    const float* __restrict__ gamma, float* __restrict__ out) {
  __shared__ unsigned short kBufS[2 * 64 * C8];  // 2 x 4KB double buffer
  __shared__ unsigned short pTile[16 * 64];      // P staging (2KB)
  const int tid = threadIdx.x;
  const int wid = tid >> 5;
  const int lane = tid & 31, half = lane >> 4, l16 = lane & 15;
  const int nBase = blockIdx.x * 16, b = blockIdx.y;
  const unsigned short* Qb = Qt + (size_t)b * NN * C8;
  const unsigned short* Kb = Kt + (size_t)b * NN * C8;
  const unsigned short* Vb = V + (size_t)b * CC * NN;
  const unsigned kBufOff = (unsigned)(size_t)(void*)kBufS;  // LDS byte offset

  // Each wave issues 4 b128 async copies (2KB) per batch; 2 waves cover 4KB.
  const unsigned lin = (unsigned)((wid * 4) * 512 + lane * 16);
#define ISSUE_K(mb, buf)                                                      \
  do {                                                                        \
    _Pragma("unroll") for (int i = 0; i < 4; ++i)                             \
        async_copy_b128(kBufOff + (unsigned)(buf) * 4096 + lin + i * 512,     \
                        Kb, (unsigned)((mb) * (C8 * 2)) + lin + i * 512);     \
  } while (0)

  // Q A-tile: row n = nBase+l16, K(channel) groups half*8..+7 / +16..+23
  const unsigned short* qp = Qb + (size_t)(nBase + l16) * C8 + half * 8;
  v16bf qa = cat(ld8(qp), ld8(qp + 16));

  v8f o_acc[8];
#pragma unroll
  for (int t = 0; t < 8; ++t) o_acc[t] = (v8f){0.f, 0.f, 0.f, 0.f, 0.f, 0.f, 0.f, 0.f};
  float rowMax[8], rowSum[8];
#pragma unroll
  for (int r = 0; r < 8; ++r) { rowMax[r] = -3.0e38f; rowSum[r] = 0.0f; }

  ISSUE_K(0, 0);  // prologue: stage first K tile

  for (int mB = 0; mB < NN; mB += 64) {
    const int buf = (mB >> 6) & 1;
    const bool more = (mB + 64) < NN;
    if (more) {
      ISSUE_K(mB + 64, buf ^ 1);  // kick next tile while we compute this one
      // prefetch next V block: 256 rows x 128B = 256 lines; 64 thr x 4
#pragma unroll
      for (int j = 0; j < 4; ++j)
        __builtin_prefetch(Vb + (size_t)(tid * 4 + j) * NN + mB + 64, 0, 1);
      asm volatile("s_wait_asynccnt 0x4" ::: "memory");  // older batch landed
    } else {
      asm volatile("s_wait_asynccnt 0x0" ::: "memory");
    }
    __syncthreads();  // all waves' async halves visible

    // ---- scores: 4 tiles of 16 keys; K-operand from LDS (ds_load_b128) ----
    const unsigned short* kbl = kBufS + buf * (64 * C8);
    v8f s[4];
#pragma unroll
    for (int t = 0; t < 4; ++t) {
      const unsigned short* kp = kbl + (t * 16 + l16) * C8 + half * 8;
      v16bf kb = cat(ld8(kp), ld8(kp + 16));
      v8f z = {};
      s[t] = wmma_bf16(qa, kb, z);
    }

    // ---- online softmax over 64 keys; row r lives on one 16-lane half ----
#pragma unroll
    for (int r = 0; r < 8; ++r) {
      float mx = fmaxf(fmaxf(s[0][r], s[1][r]), fmaxf(s[2][r], s[3][r]));
#pragma unroll
      for (int off = 8; off >= 1; off >>= 1) mx = fmaxf(mx, __shfl_xor(mx, off, 32));
      const float newMax = fmaxf(rowMax[r], mx);
      const float corr = __expf(rowMax[r] - newMax);
      rowMax[r] = newMax;
      float p[4], ls = 0.0f;
#pragma unroll
      for (int t = 0; t < 4; ++t) { p[t] = __expf(s[t][r] - newMax); ls += p[t]; }
#pragma unroll
      for (int off = 8; off >= 1; off >>= 1) ls += __shfl_xor(ls, off, 32);
      rowSum[r] = rowSum[r] * corr + ls;
#pragma unroll
      for (int t = 0; t < 8; ++t) o_acc[t][r] *= corr;
      if (wid == 0) {
#pragma unroll
        for (int t = 0; t < 4; ++t)
          pTile[(r + 8 * half) * 64 + t * 16 + l16] = f2bf_bits(p[t]);
      }
    }
    __syncthreads();

    // ---- P A-tiles (two K=32 chunks) from LDS ----
    const unsigned short* pr = pTile + l16 * 64 + half * 8;
    v16bf pa0 = cat(ld8(pr), ld8(pr + 16));
    v16bf pa1 = cat(ld8(pr + 32), ld8(pr + 48));

    // ---- O[n, c] += P * V^T ; this wave's 8 channel tiles ----
#pragma unroll
    for (int t = 0; t < 8; ++t) {
      const int c0 = (wid * 8 + t) * 16;
      const unsigned short* vp = Vb + (size_t)(c0 + l16) * NN + mB + half * 8;
      o_acc[t] = wmma_bf16(pa0, cat(ld8(vp), ld8(vp + 16)), o_acc[t]);
      o_acc[t] = wmma_bf16(pa1, cat(ld8(vp + 32), ld8(vp + 48)), o_acc[t]);
    }
  }
#undef ISSUE_K

  // ---- epilogue: out = gamma * O / rowsum + x ----
  const float g = gamma[0];
  const float* xb = x + (size_t)b * CC * NN;
  float* ob = out + (size_t)b * CC * NN;
#pragma unroll
  for (int r = 0; r < 8; ++r) {
    const int n = nBase + r + 8 * half;
    const float inv = 1.0f / rowSum[r];
#pragma unroll
    for (int t = 0; t < 8; ++t) {
      const int c = (wid * 8 + t) * 16 + l16;
      const size_t idx = (size_t)c * NN + n;
      ob[idx] = g * o_acc[t][r] * inv + xb[idx];
    }
  }
}

extern "C" void kernel_launch(void* const* d_in, const int* in_sizes, int n_in,
                              void* d_out, int out_size, void* d_ws, size_t ws_size,
                              hipStream_t stream) {
  (void)in_sizes; (void)n_in; (void)out_size; (void)ws_size;
  const float* x     = (const float*)d_in[0];
  const float* wq    = (const float*)d_in[1];
  const float* bq    = (const float*)d_in[2];
  const float* wk    = (const float*)d_in[3];
  const float* bk    = (const float*)d_in[4];
  const float* wv    = (const float*)d_in[5];
  const float* bv    = (const float*)d_in[6];
  const float* gamma = (const float*)d_in[7];
  float* out = (float*)d_out;

  // workspace layout (bf16): xt 16MB | weights 160KB | Qt 2MB | Kt 2MB | V 16MB
  unsigned short* xt  = (unsigned short*)d_ws;
  unsigned short* wqb = xt + (size_t)BB * NN * CC;
  unsigned short* wkb = wqb + C8 * CC;
  unsigned short* wvb = wkb + C8 * CC;
  unsigned short* Qt  = wvb + CC * CC;
  unsigned short* Kt  = Qt + (size_t)BB * NN * C8;
  unsigned short* Vv  = Kt + (size_t)BB * NN * C8;

  cvt_bf16<<<dim3((C8 * CC + 255) / 256), dim3(256), 0, stream>>>(wq, wqb, C8 * CC);
  cvt_bf16<<<dim3((C8 * CC + 255) / 256), dim3(256), 0, stream>>>(wk, wkb, C8 * CC);
  cvt_bf16<<<dim3((CC * CC + 255) / 256), dim3(256), 0, stream>>>(wv, wvb, CC * CC);
  transpose_x_bf16<<<dim3(NN / 32, CC / 32, BB), dim3(32, 8), 0, stream>>>(x, xt);

  proj_qk_t<<<dim3(NN / 16, C8 / 16, BB), dim3(32), 0, stream>>>(xt, wqb, bq, Qt);
  proj_qk_t<<<dim3(NN / 16, C8 / 16, BB), dim3(32), 0, stream>>>(xt, wkb, bk, Kt);
  proj_v<<<dim3(NN / 16, CC / 16, BB), dim3(32), 0, stream>>>(xt, wvb, bv, Vv);

  flash_attn<<<dim3(NN / 16, BB), dim3(64), 0, stream>>>(Qt, Kt, Vv, x, gamma, out);
}
